// GPRLayer_21045339750608
// MI455X (gfx1250) — matrix operand order repaired and checked by reading, not compiled, not run on hardware
//
#include <hip/hip_runtime.h>

// ---------------------------------------------------------------------------
// Masked linear: out = x @ (W*mask)^T + bias  on gfx1250 (MI455X, wave32, WMMA)
//   x:[1024,16384] f32, W/mask:[8192,16384] f32, bias:[8192] f32
//   out:[1024,8192] f32
// bf16 WMMA (v_wmma_f32_16x16x32_bf16), f32 accumulate.
//   - x pre-converted to bf16 into d_ws (32 MB), L2-resident thereafter.
//   - one block per 16-wide output strip -> W+mask stream HBM exactly once.
//   - 8 waves/block; wave = 8 batch tiles (128 rows) x 16 cols.
//   - pipeline: only the 8-reg bf16 B tile crosses the loop boundary;
//     f32 W/mask staging is intra-iteration; A loads in two half-batches.
// ---------------------------------------------------------------------------

typedef __attribute__((ext_vector_type(16))) __bf16    v16bf;
typedef __attribute__((ext_vector_type(8)))  __bf16    v8bf;
typedef __attribute__((ext_vector_type(8)))  float     v8f;
typedef __attribute__((ext_vector_type(4)))  float     f4;
typedef __attribute__((ext_vector_type(4)))  unsigned  u4;

#define IN_DIM  16384
#define OUT_DIM 8192
#define BATCH   1024
#define MT      8      // batch tiles (of 16 rows) per wave

union FView { f4 q[4]; float f[16]; };
union AView { v16bf v; u4 q[2]; };

// ---- prolog: f32 -> bf16 conversion of x (8 elems / thread) ----------------
__global__ void __launch_bounds__(256)
x_to_bf16(const float* __restrict__ X, __bf16* __restrict__ Xb)
{
    size_t i = ((size_t)blockIdx.x * 256u + threadIdx.x) * 8u;
    f4 a = *(const f4*)(X + i);
    f4 b = *(const f4*)(X + i + 4);
    v8bf o;
    o[0] = (__bf16)a[0]; o[1] = (__bf16)a[1];
    o[2] = (__bf16)a[2]; o[3] = (__bf16)a[3];
    o[4] = (__bf16)b[0]; o[5] = (__bf16)b[1];
    o[6] = (__bf16)b[2]; o[7] = (__bf16)b[3];
    *(v8bf*)(Xb + i) = o;
}

// ---- helpers ---------------------------------------------------------------
__device__ __forceinline__ void load16f(FView& v, const float* __restrict__ p)
{
    v.q[0] = *(const f4*)(p);
    v.q[1] = *(const f4*)(p + 4);
    v.q[2] = *(const f4*)(p + 8);
    v.q[3] = *(const f4*)(p + 12);
}

__device__ __forceinline__ v16bf fuse_cvt(const FView& w, const FView& m)
{
    v16bf b;
    #pragma unroll
    for (int j = 0; j < 16; ++j)
        b[j] = (__bf16)(w.f[j] * m.f[j]);
    return b;
}

// load 4 A tiles (half of the wave's batch tiles)
__device__ __forceinline__ void loadA4(AView av[4], const __bf16* __restrict__ xbase,
                                       int k0, int mt0)
{
    #pragma unroll
    for (int t = 0; t < 4; ++t) {
        const __bf16* xp = xbase + (size_t)((mt0 + t) * 16) * IN_DIM + k0;
        av[t].q[0] = *(const u4*)(xp);        // K chunk [lhi*16 .. +7]
        av[t].q[1] = *(const u4*)(xp + 8);    // K chunk [lhi*16+8 .. +7]
    }
}

// ---- main GEMM -------------------------------------------------------------
__global__ void __launch_bounds__(256)
masked_linear_wmma(const float*  __restrict__ W,
                   const float*  __restrict__ Mk,
                   const float*  __restrict__ bias,
                   const __bf16* __restrict__ Xb,
                   float*        __restrict__ C)
{
    const int lane  = threadIdx.x & 31;
    const int wave  = threadIdx.x >> 5;        // 0..7
    const int n0    = blockIdx.x * 16;         // output column strip
    const int mbase = wave * (MT * 16);        // 128 rows per wave

    const int l15 = lane & 15;
    const int lhi = lane >> 4;                 // 0 or 1

    // B-tile source: column n = n0+l15 of B == row (n0+l15) of W/mask,
    // K offset base = lhi*16 (lanes 0-15 hold K=0..15, lanes 16-31 K=16..31)
    const float* wrow = W  + (size_t)(n0 + l15) * IN_DIM + (lhi << 4);
    const float* mrow = Mk + (size_t)(n0 + l15) * IN_DIM + (lhi << 4);

    // A-tile source: row mbase+mt*16+l15; chunks at K offsets lhi*16 and +8
    const __bf16* xbase = Xb + (size_t)(mbase + l15) * IN_DIM + (lhi << 4);

    v8f acc[MT];
    {
        v8f z = {0.f, 0.f, 0.f, 0.f, 0.f, 0.f, 0.f, 0.f};
        #pragma unroll
        for (int i = 0; i < MT; ++i) acc[i] = z;
    }

    // ---- prime pipeline: B tile for k0 = 0 ----
    v16bf bmat;
    {
        FView wv, mv;
        load16f(wv, wrow);
        load16f(mv, mrow);
        bmat = fuse_cvt(wv, mv);
    }

    for (int k0 = 0; k0 < IN_DIM - 32; k0 += 32) {
        // first half of A tiles
        AView a1[4];
        loadA4(a1, xbase, k0, 0);

        // next iteration's B source (f32 staging lives only inside this iter)
        FView wv2, mv2;
        load16f(wv2, wrow + k0 + 32);
        load16f(mv2, mrow + k0 + 32);

        // prime L2 on the pure-HBM W/mask streams (~16 iterations ahead)
        __builtin_prefetch(wrow + k0 + 512, 0, 0);
        __builtin_prefetch(mrow + k0 + 512, 0, 0);

        #pragma unroll
        for (int mt = 0; mt < 4; ++mt)
            acc[mt] = __builtin_amdgcn_wmma_f32_16x16x32_bf16(
                          false, a1[mt].v, false, bmat,
                          (short)0, acc[mt], false, false);

        // second half of A tiles
        AView a2[4];
        loadA4(a2, xbase, k0, 4);

        #pragma unroll
        for (int mt = 4; mt < 8; ++mt)
            acc[mt] = __builtin_amdgcn_wmma_f32_16x16x32_bf16(
                          false, a2[mt - 4].v, false, bmat,
                          (short)0, acc[mt], false, false);

        // convert next B at the bottom: its loads had a full iteration in flight
        bmat = fuse_cvt(wv2, mv2);
    }

    // ---- peeled final K-step (k0 = IN_DIM-32): bmat already prepared ----
    {
        const int k0 = IN_DIM - 32;
        AView a1[4], a2[4];
        loadA4(a1, xbase, k0, 0);
        loadA4(a2, xbase, k0, 4);
        #pragma unroll
        for (int mt = 0; mt < 4; ++mt)
            acc[mt] = __builtin_amdgcn_wmma_f32_16x16x32_bf16(
                          false, a1[mt].v, false, bmat,
                          (short)0, acc[mt], false, false);
        #pragma unroll
        for (int mt = 4; mt < 8; ++mt)
            acc[mt] = __builtin_amdgcn_wmma_f32_16x16x32_bf16(
                          false, a2[mt - 4].v, false, bmat,
                          (short)0, acc[mt], false, false);
    }

    // ---- epilog: bias add + store ----
    const float bv = bias[n0 + l15];
    #pragma unroll
    for (int mt = 0; mt < MT; ++mt) {
        #pragma unroll
        for (int v = 0; v < 8; ++v) {
            const int row = mbase + mt * 16 + v + (lhi << 3);
            C[(size_t)row * OUT_DIM + n0 + l15] = acc[mt][v] + bv;
        }
    }
}

// ---------------------------------------------------------------------------
extern "C" void kernel_launch(void* const* d_in, const int* in_sizes, int n_in,
                              void* d_out, int out_size, void* d_ws, size_t ws_size,
                              hipStream_t stream)
{
    const float* x    = (const float*)d_in[0];   // [1024,16384]
    const float* w    = (const float*)d_in[1];   // [8192,16384]
    const float* bias = (const float*)d_in[2];   // [8192]
    const float* mask = (const float*)d_in[3];   // [8192,16384]
    float*       out  = (float*)d_out;           // [1024,8192]
    __bf16*      xb   = (__bf16*)d_ws;           // 32 MB scratch

    // 1) convert x to bf16 (16.7M elems, 8 per thread)
    const int conv_blocks = (BATCH * IN_DIM) / (8 * 256);   // 8192
    x_to_bf16<<<conv_blocks, 256, 0, stream>>>(x, xb);

    // 2) fused mask-multiply + WMMA GEMM + bias
    masked_linear_wmma<<<OUT_DIM / 16, 256, 0, stream>>>(w, mask, bias, xb, out);
}
